// PSBBlock_84499186581980
// MI455X (gfx1250) — matrix operand order repaired and checked by reading, not compiled, not run on hardware
//
#include <hip/hip_runtime.h>
#include <hip/hip_bf16.h>

// ---------------------------------------------------------------------------
// PSB forward for MI455X (gfx1250, wave32).
// All GEMMs use v_wmma_f32_16x16x32_f16 (f16 inputs, fp32 accumulation).
// Shapes: B=8 T=16 L=1024 D=192 N=8 F=512, BT=128, feature rows = 131072.
// ---------------------------------------------------------------------------

typedef _Float16 half8  __attribute__((ext_vector_type(8)));
typedef _Float16 half16 __attribute__((ext_vector_type(16)));
typedef float    f32x8  __attribute__((ext_vector_type(8)));

union H16u { half16 v; half8 h[2]; };

#define NT 6   // N-tiles (of 16 cols) per wave -> A fragment reused 6x

// ------------------------------ f32 -> f16 ---------------------------------
__global__ void psb_cvt_f16_kernel(const float* __restrict__ s,
                                   _Float16* __restrict__ d, int n) {
  int i = blockIdx.x * 256 + threadIdx.x;
  if (i < n) d[i] = (_Float16)s[i];
}

// ------------------------------ LayerNorm ----------------------------------
// One 64-thread block per output row of 192. mode selects input-row gather:
//   0: identity
//   1: compact from slot-padded layout: rin = (r/8)*16 + r%8   (16-row pads)
//   2: (b,t,n) -> (b,n,t) transpose:    r = (b*8+n)*16 + t, rin = (b*16+t)*8+n
__global__ void psb_ln_kernel(const float* __restrict__ x,
                              const float* __restrict__ g,
                              const float* __restrict__ bb,
                              _Float16* __restrict__ out, int mode) {
  __shared__ float red[64];
  __shared__ float stat[2];
  int r = blockIdx.x;
  int rin;
  if (mode == 0)      rin = r;
  else if (mode == 1) rin = ((r >> 3) << 4) + (r & 7);
  else { int bn = r >> 4, t = r & 15, b = bn >> 3, n = bn & 7;
         rin = ((b << 4) + t) * 8 + n; }
  const float* xr = x + (size_t)rin * 192;
  int t = threadIdx.x;
  float v0 = xr[t], v1 = xr[t + 64], v2 = xr[t + 128];
  red[t] = v0 + v1 + v2;
  __syncthreads();
  for (int s = 32; s > 0; s >>= 1) { if (t < s) red[t] += red[t + s]; __syncthreads(); }
  if (t == 0) stat[0] = red[0] * (1.0f / 192.0f);
  __syncthreads();
  float mean = stat[0];
  float d0 = v0 - mean, d1 = v1 - mean, d2 = v2 - mean;
  red[t] = d0 * d0 + d1 * d1 + d2 * d2;
  __syncthreads();
  for (int s = 32; s > 0; s >>= 1) { if (t < s) red[t] += red[t + s]; __syncthreads(); }
  if (t == 0) stat[1] = rsqrtf(red[0] * (1.0f / 192.0f) + 1e-5f);
  __syncthreads();
  float rstd = stat[1];
  _Float16* o = out + (size_t)r * 192;
  o[t]       = (_Float16)(d0 * rstd * g[t]       + bb[t]);
  o[t + 64]  = (_Float16)(d1 * rstd * g[t + 64]  + bb[t + 64]);
  o[t + 128] = (_Float16)(d2 * rstd * g[t + 128] + bb[t + 128]);
}

// ------------------------------ WMMA GEMM ----------------------------------
// C[M,N] = scale * (A[M,K] @ W[N,K]^T) + bias (+resid), optional ReLU.
// Batched via element strides. One wave computes a 16-row strip of up to
// NT 16x16 tiles. store_mode: 0 normal row-major (ldc=N)
//                             1 slot-pad: row -> (row/8)*16 + row%8
//                             2 V-transpose: (bt,l,d) -> (bt,d,l), L=1024,D=192
//
// Fast path (nloc == NT): all NT B-fragments + next A-fragment are issued
// before the WMMA chain, so loads stay in flight behind the matrix pipe
// (incremental s_wait_loadcnt instead of a full stall per tile) and the hot
// loop is branch-free.
__global__ __launch_bounds__(128) void psb_gemm_wmma_kernel(
    const _Float16* __restrict__ A, const _Float16* __restrict__ Bw,
    const float* __restrict__ bias, const float* __restrict__ resid,
    float* __restrict__ Cf, _Float16* __restrict__ Ch,
    int M, int N, int K,
    long strideA, long strideB, long strideC,
    float out_scale, int store_mode, int relu) {
  int lane = threadIdx.x;                               // 0..31
  int mt = blockIdx.y * blockDim.y + threadIdx.y;       // wave-uniform
  if (mt * 16 >= M) return;
  int batch = blockIdx.z;
  int ntiles = N >> 4;
  int nt0 = blockIdx.x * NT;
  int nloc = ntiles - nt0; if (nloc > NT) nloc = NT;    // wave-uniform
  if (nloc <= 0) return;
  const _Float16* Ab = A + (size_t)batch * strideA;
  const _Float16* Bb = Bw + (size_t)batch * strideB;
  int m = lane & 15, hi = lane >> 4;

  f32x8 acc[NT];
#pragma unroll
  for (int j = 0; j < NT; ++j)
#pragma unroll
    for (int e = 0; e < 8; ++e) acc[j][e] = 0.0f;

  const _Float16* arow = Ab + (size_t)(mt * 16 + m) * K + hi * 8;
  // Per-tile B row base pointers (hoisted out of the k loop).
  const _Float16* brow[NT];
#pragma unroll
  for (int j = 0; j < NT; ++j)
    brow[j] = Bb + (size_t)((nt0 + (j < nloc ? j : 0)) * 16 + m) * K + hi * 8;

  if (nloc == NT) {
    // ---------------- branch-free pipelined path ----------------
    H16u a;
    a.h[0] = *(const half8*)(arow);
    a.h[1] = *(const half8*)(arow + 16);
    for (int k0 = 0; k0 < K; k0 += 32) {
      H16u b[NT];
#pragma unroll
      for (int j = 0; j < NT; ++j) {
        b[j].h[0] = *(const half8*)(brow[j] + k0);
        b[j].h[1] = *(const half8*)(brow[j] + k0 + 16);
      }
      H16u an;
      int kn = k0 + 32;
      if (kn < K) {                      // prefetch next A fragment
        an.h[0] = *(const half8*)(arow + kn);
        an.h[1] = *(const half8*)(arow + kn + 16);
      }
#pragma unroll
      for (int j = 0; j < NT; ++j)
        acc[j] = __builtin_amdgcn_wmma_f32_16x16x32_f16(
            false, a.v, false, b[j].v, (short)0, acc[j], false, false);
      if (kn < K) a = an;
    }
  } else {
    // ---------------- ragged remainder path ----------------
    for (int k0 = 0; k0 < K; k0 += 32) {
      H16u a;
      a.h[0] = *(const half8*)(arow + k0);
      a.h[1] = *(const half8*)(arow + k0 + 16);
      H16u b[NT];
#pragma unroll
      for (int j = 0; j < NT; ++j) {
        if (j >= nloc) break;
        b[j].h[0] = *(const half8*)(brow[j] + k0);
        b[j].h[1] = *(const half8*)(brow[j] + k0 + 16);
      }
#pragma unroll
      for (int j = 0; j < NT; ++j) {
        if (j >= nloc) break;
        acc[j] = __builtin_amdgcn_wmma_f32_16x16x32_f16(
            false, a.v, false, b[j].v, (short)0, acc[j], false, false);
      }
    }
  }

#pragma unroll
  for (int j = 0; j < NT; ++j) {
    if (j >= nloc) break;
    int col = (nt0 + j) * 16 + m;
#pragma unroll
    for (int rI = 0; rI < 8; ++rI) {
      int row = mt * 16 + (hi ? 8 + rI : rI);  // C layout: vgpr r -> M=r / M=8+r
      float v = acc[j][rI] * out_scale;
      if (bias)  v += bias[col];
      if (relu)  v = fmaxf(v, 0.0f);
      size_t idx;
      if (store_mode == 0)
        idx = (size_t)batch * strideC + (size_t)row * N + col;
      else if (store_mode == 1)
        idx = (size_t)(((row >> 3) << 4) + (row & 7)) * N + col;
      else
        idx = (size_t)(row >> 10) * (192 * 1024) + (size_t)col * 1024 + (row & 1023);
      if (resid) v += resid[idx];
      if (Cf) Cf[idx] = v;
      if (Ch) Ch[idx] = (_Float16)v;
    }
  }
}

// ---------------- inverse attention: softmax over slots (n) ----------------
// attn (BT,16,1024) f32, rows 0..7 valid; in-place column softmax over n.
__global__ void psb_softmax_slots_kernel(float* __restrict__ attn) {
  int id = blockIdx.x * 256 + threadIdx.x;        // 0..131071 = bt*1024 + l
  int bt = id >> 10, l = id & 1023;
  float* base = attn + (size_t)bt * 16 * 1024 + l;
  float a[8], mx = -1e30f;
#pragma unroll
  for (int n = 0; n < 8; ++n) { a[n] = base[n * 1024]; mx = fmaxf(mx, a[n]); }
  float s = 0.0f;
#pragma unroll
  for (int n = 0; n < 8; ++n) { a[n] = expf(a[n] - mx); s += a[n]; }
  float inv = 1.0f / s;
#pragma unroll
  for (int n = 0; n < 8; ++n) base[n * 1024] = a[n] * inv;
}

// ------------- inverse attention: renormalize over keys (l) ----------------
// One 256-thread block per (bt,n) row; writes f16 weights (padded buffer,
// pad rows pre-zeroed by memset).
__global__ void psb_renorm_keys_kernel(const float* __restrict__ attn,
                                       _Float16* __restrict__ wh) {
  __shared__ float red[256];
  __shared__ float stot;
  int row = blockIdx.x;                 // bt*8 + n
  int bt = row >> 3, n = row & 7;
  const float* p = attn + (size_t)(bt * 16 + n) * 1024;
  int t = threadIdx.x;
  red[t] = p[t] + p[t + 256] + p[t + 512] + p[t + 768];
  __syncthreads();
  for (int s = 128; s > 0; s >>= 1) { if (t < s) red[t] += red[t + s]; __syncthreads(); }
  if (t == 0) stot = 1.0f / red[0];
  __syncthreads();
  float inv = stot;
  _Float16* o = wh + (size_t)(bt * 16 + n) * 1024;
#pragma unroll
  for (int k = 0; k < 4; ++k)
    o[t + k * 256] = (_Float16)(p[t + k * 256] * inv);
}

// --------------------- small 4-head MHA (time / object) --------------------
// qkv (Lseq*Bb, 576) f32 rows ordered (l*Bb + b). grid (Lseq, Bb, 4),
// block = Lseq threads. out (Lseq*Bb, 192) f16.
__global__ void psb_mha_kernel(const float* __restrict__ qkv,
                               _Float16* __restrict__ out,
                               int Lseq, int Bb, float scale) {
  __shared__ float sc[128];
  __shared__ float red2[2];
  int l = blockIdx.x, b = blockIdx.y, h = blockIdx.z;
  int t = threadIdx.x;                  // key index s
  const float* qrow = qkv + ((size_t)l * Bb + b) * 576 + h * 48;
  const float* krow = qkv + ((size_t)t * Bb + b) * 576 + 192 + h * 48;
  float dot = 0.0f;
#pragma unroll
  for (int e = 0; e < 48; ++e) dot += qrow[e] * krow[e];
  sc[t] = dot * scale;
  __syncthreads();
  if (t == 0) { float m = -1e30f; for (int i = 0; i < Lseq; ++i) m = fmaxf(m, sc[i]); red2[0] = m; }
  __syncthreads();
  float e = expf(sc[t] - red2[0]);
  sc[t] = e;
  __syncthreads();
  if (t == 0) { float s = 0.0f; for (int i = 0; i < Lseq; ++i) s += sc[i]; red2[1] = 1.0f / s; }
  __syncthreads();
  float p = sc[t] * red2[1];
  sc[t] = p;
  __syncthreads();
  if (t < 48) {
    float acc = 0.0f;
    for (int s = 0; s < Lseq; ++s)
      acc += sc[s] * qkv[((size_t)s * Bb + b) * 576 + 384 + h * 48 + t];
    out[((size_t)l * Bb + b) * 192 + h * 48 + t] = (_Float16)acc;
  }
}

// ---------------------------------------------------------------------------
extern "C" void kernel_launch(void* const* d_in, const int* in_sizes, int n_in,
                              void* d_out, int out_size, void* d_ws, size_t ws_size,
                              hipStream_t stream) {
  (void)in_sizes; (void)n_in; (void)out_size; (void)ws_size;
  const float* features = (const float*)d_in[0];
  const float* prev     = (const float*)d_in[1];
  const float* lnq_g  = (const float*)d_in[2];  const float* lnq_b  = (const float*)d_in[3];
  const float* lnkv_g = (const float*)d_in[4];  const float* lnkv_b = (const float*)d_in[5];
  const float* wq = (const float*)d_in[6];  const float* wk = (const float*)d_in[7];
  const float* wv = (const float*)d_in[8];  const float* wo = (const float*)d_in[9];
  const float* lnt_g = (const float*)d_in[10]; const float* lnt_b = (const float*)d_in[11];
  const float* t_in_w = (const float*)d_in[12]; const float* t_in_b = (const float*)d_in[13];
  const float* t_out_w = (const float*)d_in[14]; const float* t_out_b = (const float*)d_in[15];
  const float* lno_g = (const float*)d_in[16]; const float* lno_b = (const float*)d_in[17];
  const float* o_in_w = (const float*)d_in[18]; const float* o_in_b = (const float*)d_in[19];
  const float* o_out_w = (const float*)d_in[20]; const float* o_out_b = (const float*)d_in[21];
  const float* lnp_g = (const float*)d_in[22]; const float* lnp_b = (const float*)d_in[23];
  const float* w1 = (const float*)d_in[24]; const float* b1 = (const float*)d_in[25];
  const float* w2 = (const float*)d_in[26]; const float* b2 = (const float*)d_in[27];
  float* out = (float*)d_out;

  const int Bc = 8, Tc = 16, Lc = 1024, Dc = 192, Nc = 8, Fc = 512;
  const int BT = Bc * Tc;                 // 128
  const int MFEAT = BT * Lc;              // 131072 feature rows

  // ---------------- workspace layout (persistent region) ----------------
  char* w = (char*)d_ws; size_t off = 0;
  auto alloc = [&](size_t bytes) { char* p = w + off; off += (bytes + 255) & ~255ull; return p; };
  _Float16* xkv_h   = (_Float16*)alloc((size_t)MFEAT * Dc * 2);   // 50.3 MB, recycled below
  _Float16* slots_h = (_Float16*)alloc((size_t)1024 * Dc * 2);
  _Float16* K_h     = (_Float16*)alloc((size_t)MFEAT * Dc * 2);   // (BT,L,D)
  _Float16* V_t     = (_Float16*)alloc((size_t)MFEAT * Dc * 2);   // (BT,D,L)
  _Float16* wq_h  = (_Float16*)alloc(36864 * 2);
  _Float16* wk_h  = (_Float16*)alloc(36864 * 2);
  _Float16* wv_h  = (_Float16*)alloc(36864 * 2);
  _Float16* wo_h  = (_Float16*)alloc(36864 * 2);
  _Float16* tin_h = (_Float16*)alloc(110592 * 2);
  _Float16* tout_h= (_Float16*)alloc(36864 * 2);
  _Float16* oin_h = (_Float16*)alloc(110592 * 2);
  _Float16* oout_h= (_Float16*)alloc(36864 * 2);
  _Float16* w1_h  = (_Float16*)alloc(98304 * 2);
  _Float16* w2_h  = (_Float16*)alloc(98304 * 2);

  // Recycled region carved out of xkv_h (valid once K/V GEMMs have been
  // issued; stream ordering guarantees safety). Total 25.0 MB < 50.3 MB.
  char* rw = (char*)xkv_h; size_t roff = 0;
  auto sub = [&](size_t bytes) { char* p = rw + roff; roff += (bytes + 255) & ~255ull; return p; };
  _Float16* q_h    = (_Float16*)sub((size_t)BT * 16 * Dc * 2);    // slot-padded
  float*    attn   = (float*)   sub((size_t)BT * 16 * Lc * 4);
  _Float16* w_h    = (_Float16*)sub((size_t)BT * 16 * Lc * 2);    // slot-padded
  _Float16* s1_h   = (_Float16*)sub((size_t)BT * 16 * Dc * 2);
  float*    s2     = (float*)   sub((size_t)BT * 16 * Dc * 4);
  _Float16* st_h   = (_Float16*)sub((size_t)1024 * Dc * 2);
  float*    tqkv   = (float*)   sub((size_t)1024 * 576 * 4);
  _Float16* tatt_h = (_Float16*)sub((size_t)1024 * Dc * 2);
  float*    s3     = (float*)   sub((size_t)1024 * Dc * 4);
  _Float16* so_h   = (_Float16*)sub((size_t)1024 * Dc * 2);
  float*    oqkv   = (float*)   sub((size_t)1024 * 576 * 4);
  _Float16* oatt_h = (_Float16*)sub((size_t)1024 * Dc * 2);
  float*    s4     = (float*)   sub((size_t)1024 * Dc * 4);
  _Float16* s4ln_h = (_Float16*)sub((size_t)1024 * Dc * 2);
  _Float16* h1_h   = (_Float16*)sub((size_t)1024 * Fc * 2);

  auto cvt = [&](const float* s, _Float16* d, int n) {
    psb_cvt_f16_kernel<<<dim3((n + 255) / 256), dim3(256), 0, stream>>>(s, d, n);
  };
  auto gemm = [&](const _Float16* A, const _Float16* Bw, const float* bias,
                  const float* resid, float* Cf, _Float16* Ch,
                  int M, int N, int K, long sA, long sB, long sC, int batch,
                  float scl, int smode, int relu, int yw) {
    dim3 grid((N / 16 + NT - 1) / NT, (M / 16 + yw - 1) / yw, batch);
    dim3 block(32, yw);
    psb_gemm_wmma_kernel<<<grid, block, 0, stream>>>(
        A, Bw, bias, resid, Cf, Ch, M, N, K, sA, sB, sC, scl, smode, relu);
  };

  // 0) weight conversions to f16
  cvt(wq, wq_h, 36864);   cvt(wk, wk_h, 36864);
  cvt(wv, wv_h, 36864);   cvt(wo, wo_h, 36864);
  cvt(t_in_w, tin_h, 110592); cvt(t_out_w, tout_h, 36864);
  cvt(o_in_w, oin_h, 110592); cvt(o_out_w, oout_h, 36864);
  cvt(w1, w1_h, 98304);   cvt(w2, w2_h, 98304);

  // 1) LayerNorms of inputs -> f16
  psb_ln_kernel<<<dim3(MFEAT), dim3(64), 0, stream>>>(features, lnkv_g, lnkv_b, xkv_h, 0);
  psb_ln_kernel<<<dim3(1024),  dim3(64), 0, stream>>>(prev, lnq_g, lnq_b, slots_h, 0);

  // 2) K = xkv @ wk^T (f16, (BT,L,D)); V stored transposed (BT,D,L)
  gemm(xkv_h, wk_h, nullptr, nullptr, nullptr, K_h, MFEAT, Dc, Dc, 0, 0, 0, 1, 1.0f, 0, 0, 4);
  gemm(xkv_h, wv_h, nullptr, nullptr, nullptr, V_t, MFEAT, Dc, Dc, 0, 0, 0, 1, 1.0f, 2, 0, 4);

  // zero the slot-padded buffers (after KV GEMMs: they alias xkv_h region)
  hipMemsetAsync(q_h, 0, (size_t)BT * 16 * Dc * 2, stream);
  hipMemsetAsync(w_h, 0, (size_t)BT * 16 * Lc * 2, stream);

  // 3) q = (slots @ wq^T) * D^-0.5, stored slot-padded (BT,16,D) f16
  gemm(slots_h, wq_h, nullptr, nullptr, nullptr, q_h, 1024, Dc, Dc, 0, 0, 0, 1,
       0.072168783649f /* 192^-0.5 */, 1, 0, 4);

  // 4) attn = q @ K^T, batched over BT -> (BT,16,L) f32
  gemm(q_h, K_h, nullptr, nullptr, attn, nullptr, 16, Lc, Dc,
       16L * Dc, (long)Lc * Dc, 16L * Lc, BT, 1.0f, 0, 0, 1);

  // 5) softmax over slots, then renormalize over keys -> f16 weights
  psb_softmax_slots_kernel<<<dim3(BT * Lc / 256), dim3(256), 0, stream>>>(attn);
  psb_renorm_keys_kernel<<<dim3(BT * Nc), dim3(256), 0, stream>>>(attn, w_h);

  // 6) s = w @ V  (batched; B operand is V_t so fragments are contiguous)
  gemm(w_h, V_t, nullptr, nullptr, nullptr, s1_h, 16, Dc, Lc,
       16L * Lc, (long)Dc * Lc, 0, BT, 1.0f, 0, 0, 1);

  // 7) s = s @ wo^T (pad rows are zero and stay zero)
  gemm(s1_h, wo_h, nullptr, nullptr, s2, nullptr, BT * 16, Dc, Dc, 0, 0, 0, 1, 1.0f, 0, 0, 4);

  // 8) time attention: LN (compact pad) -> qkv -> MHA(L=128,B=8) -> out proj
  psb_ln_kernel<<<dim3(1024), dim3(64), 0, stream>>>(s2, lnt_g, lnt_b, st_h, 1);
  gemm(st_h, tin_h, t_in_b, nullptr, tqkv, nullptr, 1024, 576, Dc, 0, 0, 0, 1, 1.0f, 0, 0, 4);
  psb_mha_kernel<<<dim3(128, 8, 4), dim3(128), 0, stream>>>(tqkv, tatt_h, 128, 8,
                                                            0.14433756729f /* 48^-0.5 */);
  gemm(tatt_h, tout_h, t_out_b, nullptr, s3, nullptr, 1024, Dc, Dc, 0, 0, 0, 1, 1.0f, 0, 0, 4);

  // 9) object attention: LN with (b,t,n)->(b,n,t) gather -> qkv -> MHA(L=64,B=16)
  psb_ln_kernel<<<dim3(1024), dim3(64), 0, stream>>>(s3, lno_g, lno_b, so_h, 2);
  gemm(so_h, oin_h, o_in_b, nullptr, oqkv, nullptr, 1024, 576, Dc, 0, 0, 0, 1, 1.0f, 0, 0, 4);
  psb_mha_kernel<<<dim3(64, 16, 4), dim3(64), 0, stream>>>(oqkv, oatt_h, 64, 16,
                                                           0.14433756729f);
  gemm(oatt_h, oout_h, o_out_b, nullptr, s4, nullptr, 1024, Dc, Dc, 0, 0, 0, 1, 1.0f, 0, 0, 4);

  // 10) final LN + FFN + residual (flat reinterpretation matches prev_slots)
  psb_ln_kernel<<<dim3(1024), dim3(64), 0, stream>>>(s4, lnp_g, lnp_b, s4ln_h, 0);
  gemm(s4ln_h, w1_h, b1, nullptr, nullptr, h1_h, 1024, Fc, Dc, 0, 0, 0, 1, 1.0f, 0, 1, 4);
  gemm(h1_h, w2_h, b2, prev, out, nullptr, 1024, Dc, Fc, 0, 0, 0, 1, 1.0f, 0, 0, 4);
}
